// Net_5738076307526
// MI455X (gfx1250) — matrix operand order repaired and checked by reading, not compiled, not run on hardware
//
#include <hip/hip_runtime.h>

// ---------------------------------------------------------------------------
// CDNA5 (gfx1250, wave32) edge-attention block.
//   * f16 WMMA (v_wmma_f32_16x16x32_f16), f32 accumulate
//   * 128-thread blocks = 4 wave32 waves; each wave owns 16 rows x 128 cols
//   * agts/ctx pre-converted to f16 once; gathered A-fragments are fed to
//     WMMA directly from global (per-lane row pointers), no LDS round trip
//   * GroupNorm computed in C-fragment registers via 16-lane xor-shuffles
//   * scatter-add via native global_atomic_add_f32 (unsafeAtomicAdd)
// ---------------------------------------------------------------------------

typedef __attribute__((ext_vector_type(8)))  _Float16 v8h;
typedef __attribute__((ext_vector_type(16))) _Float16 v16h;
typedef __attribute__((ext_vector_type(8)))  float    v8f;

#define DD 128         // feature dim
#define BM 64          // rows (edges / agents) per block
#define TPB 128        // 4 waves of 32

__device__ __forceinline__ v8f vzero8f() {
  v8f z = {0.f, 0.f, 0.f, 0.f, 0.f, 0.f, 0.f, 0.f};
  return z;
}

__device__ __forceinline__ v16h cat8(v8h lo, v8h hi) {
  return __builtin_shufflevector(lo, hi, 0, 1, 2, 3, 4, 5, 6, 7,
                                 8, 9, 10, 11, 12, 13, 14, 15);
}

// A fragment (16x32 f16) from an LDS tile stored row-major [16][128].
// Lane l holds row m=l&15; halves 0..7 are K = 8*(l>>4)+0..7, halves 8..15
// are K = 16+8*(l>>4)+0..7 -> two contiguous 16B chunks per lane.
__device__ __forceinline__ v16h a_frag_lds(const _Float16* buf, int lane, int kbase) {
  const _Float16* p = buf + (lane & 15) * DD + kbase + ((lane >> 4) << 3);
  return cat8(*(const v8h*)(p), *(const v8h*)(p + 16));
}

// A fragment from a per-lane row pointer (already offset by row*DD + khalf);
// used to feed WMMA straight from (gathered) global f16 rows.
__device__ __forceinline__ v16h a_frag_ptr(const _Float16* prow, int kbase) {
  return cat8(*(const v8h*)(prow + kbase), *(const v8h*)(prow + kbase + 16));
}

// B fragment (32x16 f16) from a row-major [K][128] f16 weight matrix.
// Lane l holds column n = nbase + (l&15); elements 0..7 are K = 8*(l>>4)+0..7,
// elements 8..15 are K = 16+8*(l>>4)+0..7 (same order as the A fragment).
__device__ __forceinline__ v16h b_frag(const _Float16* W, int kbase, int nbase, int lane) {
  const _Float16* p = W + (size_t)(kbase + ((lane >> 4) << 3)) * DD + nbase + (lane & 15);
  v16h b;
#pragma unroll
  for (int s = 0; s < 8; ++s) {
    b[s] = p[s * DD];
    b[s + 8] = p[(s + 16) * DD];
  }
  return b;
}

// acc[nt] covers columns nt*16 .. nt*16+15 of this wave's 16 rows (C layout:
// VGPR r <-> row r + 8*(lane>>4), column nt*16 + (lane&15)).
__device__ __forceinline__ void gemm128_lds(v8f acc[8], const _Float16* abuf,
                                            const _Float16* W, int wk0, int lane) {
#pragma unroll
  for (int kc = 0; kc < 4; ++kc) {
    v16h a = a_frag_lds(abuf, lane, kc * 32);
#pragma unroll
    for (int nt = 0; nt < 8; ++nt) {
      v16h b = b_frag(W, wk0 + kc * 32, nt * 16, lane);
      acc[nt] = __builtin_amdgcn_wmma_f32_16x16x32_f16(
          false, a, false, b, (short)0, acc[nt], false, false);
    }
  }
}

__device__ __forceinline__ void gemm128_g(v8f acc[8], const _Float16* prow,
                                          const _Float16* W, int wk0, int lane) {
#pragma unroll
  for (int kc = 0; kc < 4; ++kc) {
    v16h a = a_frag_ptr(prow, kc * 32);
#pragma unroll
    for (int nt = 0; nt < 8; ++nt) {
      v16h b = b_frag(W, wk0 + kc * 32, nt * 16, lane);
      acc[nt] = __builtin_amdgcn_wmma_f32_16x16x32_f16(
          false, a, false, b, (short)0, acc[nt], false, false);
    }
  }
}

__device__ __forceinline__ void add_bias(v8f acc[8], const float* b, int lane) {
  const int nl = lane & 15;
#pragma unroll
  for (int nt = 0; nt < 8; ++nt) {
    float bb = b[nt * 16 + nl];
#pragma unroll
    for (int r = 0; r < 8; ++r) acc[nt][r] += bb;
  }
}

// Per-row GroupNorm(1, 128) inside the C fragments: row values for VGPR r live
// on the 16 lanes sharing (lane>>4); xor-shuffles 1/2/4/8 reduce within them.
__device__ __forceinline__ void group_norm_rows(v8f acc[8], const float* gamma,
                                                const float* beta, int lane,
                                                bool relu) {
  const int nl = lane & 15;
  float gv[8], bv[8];
#pragma unroll
  for (int nt = 0; nt < 8; ++nt) {
    gv[nt] = gamma[nt * 16 + nl];
    bv[nt] = beta[nt * 16 + nl];
  }
#pragma unroll
  for (int r = 0; r < 8; ++r) {
    float s = 0.f, ss = 0.f;
#pragma unroll
    for (int nt = 0; nt < 8; ++nt) {
      float v = acc[nt][r];
      s += v;
      ss += v * v;
    }
#pragma unroll
    for (int off = 1; off < 16; off <<= 1) {
      s  += __shfl_xor(s,  off, 32);
      ss += __shfl_xor(ss, off, 32);
    }
    float mu  = s * (1.f / 128.f);
    float var = ss * (1.f / 128.f) - mu * mu;
    float rs  = rsqrtf(var + 1e-5f);
#pragma unroll
    for (int nt = 0; nt < 8; ++nt) {
      float v = (acc[nt][r] - mu) * rs * gv[nt] + bv[nt];
      if (relu) v = fmaxf(v, 0.f);
      acc[nt][r] = v;
    }
  }
}

// Store C fragments as f16 into an LDS tile [16][128] (row-major).
__device__ __forceinline__ void store_tile_f16(_Float16* buf, const v8f acc[8], int lane) {
  const int mh = (lane >> 4) << 3;
  const int nl = lane & 15;
#pragma unroll
  for (int r = 0; r < 8; ++r) {
    int m = r + mh;
#pragma unroll
    for (int nt = 0; nt < 8; ++nt)
      buf[m * DD + nt * 16 + nl] = (_Float16)acc[nt][r];
  }
}

// ---------------------------------------------------------------------------
// Kernel: f32 -> f16 conversion (weights and activation tables)
// ---------------------------------------------------------------------------
__global__ void convert_f16_kernel(const float* __restrict__ src,
                                   _Float16* __restrict__ dst, int n) {
  int i = blockIdx.x * 256 + threadIdx.x;
  if (i < n) dst[i] = (_Float16)src[i];
}

// ---------------------------------------------------------------------------
// Kernel: abuf = agts @ W_a  -- A straight from global f16, no LDS at all
// ---------------------------------------------------------------------------
__global__ __launch_bounds__(TPB) void base_kernel(const _Float16* __restrict__ agtsh,
                                                   const _Float16* __restrict__ W_ah,
                                                   float* __restrict__ abuf, int nrows) {
  const int tid = threadIdx.x, lane = tid & 31, w = tid >> 5;
  const long r0 = (long)blockIdx.x * BM;
  const int khalf = (lane >> 4) << 3;

  long arow = r0 + w * 16 + (lane & 15);
  if (arow >= nrows) arow = nrows - 1;  // clamp (grid is exact anyway)
  const _Float16* pa = agtsh + (size_t)arow * DD + khalf;

  v8f acc[8];
#pragma unroll
  for (int nt = 0; nt < 8; ++nt) acc[nt] = vzero8f();
  gemm128_g(acc, pa, W_ah, 0, lane);

  const int mh = (lane >> 4) << 3, nl = lane & 15;
#pragma unroll
  for (int r = 0; r < 8; ++r) {
    long m = r0 + w * 16 + r + mh;
    if (m < nrows) {
#pragma unroll
      for (int nt = 0; nt < 8; ++nt)
        abuf[m * DD + nt * 16 + nl] = acc[nt][r];
    }
  }
}

// ---------------------------------------------------------------------------
// Kernel: per-edge pipeline (4 chained GEMMs) + atomic scatter into abuf
// ---------------------------------------------------------------------------
__global__ __launch_bounds__(TPB) void edge_kernel(
    const _Float16* __restrict__ agtsh, const _Float16* __restrict__ ctxh,
    const float* __restrict__ agt_ctrs, const float* __restrict__ ctx_ctrs,
    const int* __restrict__ hi, const int* __restrict__ wi, int E,
    const float* __restrict__ W_d0, const float* __restrict__ b_d0,
    const float* __restrict__ b_d1, const float* __restrict__ g_d1,
    const float* __restrict__ be_d1,
    const float* __restrict__ b_q, const float* __restrict__ g_q,
    const float* __restrict__ be_q,
    const float* __restrict__ b_c, const float* __restrict__ g_c,
    const float* __restrict__ be_c,
    const _Float16* __restrict__ W_d1h, const _Float16* __restrict__ W_qh,
    const _Float16* __restrict__ W_ch, const _Float16* __restrict__ W_c2h,
    float* __restrict__ abuf) {
  __shared__ _Float16 buf0[BM * DD];   // d0, later h
  __shared__ _Float16 bufD[BM * DD];   // d
  __shared__ _Float16 bufQ[BM * DD];   // q
  __shared__ int   ehi[BM], ewi[BM];
  __shared__ float dxs[BM], dys[BM];

  const int tid = threadIdx.x, lane = tid & 31, w = tid >> 5;
  const long e0 = (long)blockIdx.x * BM;

  if (tid < BM) {
    long e = e0 + tid;
    int h = (e < E) ? hi[e] : 0;
    int c = (e < E) ? wi[e] : 0;
    ehi[tid] = h;
    ewi[tid] = c;
    dxs[tid] = agt_ctrs[h * 2 + 0] - ctx_ctrs[c * 2 + 0];
    dys[tid] = agt_ctrs[h * 2 + 1] - ctx_ctrs[c * 2 + 1];
  }
  __syncthreads();

  // Per-lane gathered A-row pointers (f16 rows, offset by khalf).
  const int khalf = (lane >> 4) << 3;
  const int mrow  = w * 16 + (lane & 15);
  const _Float16* paA = agtsh + (size_t)ehi[mrow] * DD + khalf;  // agts[hi] row
  const _Float16* paC = ctxh  + (size_t)ewi[mrow] * DD + khalf;  // ctx[wi]  row

  // ---- stage 0: d0 = relu([dx,dy] @ W_d0 + b_d0) -> buf0 (f32 math) ----
  for (int i = tid; i < BM * DD; i += TPB) {
    int e = i >> 7, c = i & 127;
    float v = dxs[e] * W_d0[c] + dys[e] * W_d0[DD + c] + b_d0[c];
    buf0[i] = (_Float16)fmaxf(v, 0.f);
  }
  __syncthreads();

  _Float16* myb0 = buf0 + w * 16 * DD;
  _Float16* mybD = bufD + w * 16 * DD;
  _Float16* mybQ = bufQ + w * 16 * DD;
  v8f acc[8];

  // ---- stage 1: d = relu(GN(d0 @ W_d1 + b_d1)) ----
#pragma unroll
  for (int nt = 0; nt < 8; ++nt) acc[nt] = vzero8f();
  gemm128_lds(acc, myb0, W_d1h, 0, lane);
  add_bias(acc, b_d1, lane);
  group_norm_rows(acc, g_d1, be_d1, lane, true);
  store_tile_f16(mybD, acc, lane);

  // ---- stage 2: q = relu(GN(agts[hi] @ W_q + b_q)), A direct from global ----
#pragma unroll
  for (int nt = 0; nt < 8; ++nt) acc[nt] = vzero8f();
  gemm128_g(acc, paA, W_qh, 0, lane);
  add_bias(acc, b_q, lane);
  group_norm_rows(acc, g_q, be_q, lane, true);
  store_tile_f16(mybQ, acc, lane);

  // ---- stage 3: h = relu(GN(concat(d,q,ctx) @ W_c + b_c)) ----
#pragma unroll
  for (int nt = 0; nt < 8; ++nt) acc[nt] = vzero8f();
  gemm128_lds(acc, mybD, W_ch,   0, lane);   // d   : W_c rows   0..127
  gemm128_lds(acc, mybQ, W_ch, 128, lane);   // q   : W_c rows 128..255
  gemm128_g  (acc, paC,  W_ch, 256, lane);   // ctx : W_c rows 256..383
  add_bias(acc, b_c, lane);
  group_norm_rows(acc, g_c, be_c, lane, true);
  __syncthreads();
  store_tile_f16(myb0, acc, lane);           // h overwrites d0
  __syncthreads();

  // ---- stage 4: h2 = h @ W_c2 ; abuf[hi[e]] += h2[e] ----
#pragma unroll
  for (int nt = 0; nt < 8; ++nt) acc[nt] = vzero8f();
  gemm128_lds(acc, myb0, W_c2h, 0, lane);

  const int mh = (lane >> 4) << 3, nl = lane & 15;
#pragma unroll
  for (int r = 0; r < 8; ++r) {
    int ml = w * 16 + r + mh;
    if (e0 + ml < E) {
      long row = (long)ehi[ml];
#pragma unroll
      for (int nt = 0; nt < 8; ++nt)
        unsafeAtomicAdd(&abuf[row * DD + nt * 16 + nl], acc[nt][r]);
    }
  }
}

// ---------------------------------------------------------------------------
// Kernel: a = relu(GN(abuf)); a = GN(a @ W_l + b_l); out = relu(a + agts)
// ---------------------------------------------------------------------------
__global__ __launch_bounds__(TPB) void final_kernel(
    const float* __restrict__ abuf, const float* __restrict__ agts,
    const float* __restrict__ g_n, const float* __restrict__ be_n,
    const _Float16* __restrict__ W_lh, const float* __restrict__ b_l,
    const float* __restrict__ g_l, const float* __restrict__ be_l,
    float* __restrict__ out, int nrows) {
  __shared__ _Float16 buf[BM * DD];
  const int tid = threadIdx.x, lane = tid & 31, w = tid >> 5;
  const long r0 = (long)blockIdx.x * BM;
  const int mh = (lane >> 4) << 3, nl = lane & 15;

  // Load abuf rows directly into C-fragment layout.
  v8f acc[8];
#pragma unroll
  for (int r = 0; r < 8; ++r) {
    long m = r0 + w * 16 + r + mh;
#pragma unroll
    for (int nt = 0; nt < 8; ++nt)
      acc[nt][r] = (m < nrows) ? abuf[m * DD + nt * 16 + nl] : 0.f;
  }

  group_norm_rows(acc, g_n, be_n, lane, true);
  store_tile_f16(buf + w * 16 * DD, acc, lane);
  __syncthreads();

#pragma unroll
  for (int nt = 0; nt < 8; ++nt) acc[nt] = vzero8f();
  gemm128_lds(acc, buf + w * 16 * DD, W_lh, 0, lane);
  add_bias(acc, b_l, lane);
  group_norm_rows(acc, g_l, be_l, lane, false);

#pragma unroll
  for (int r = 0; r < 8; ++r) {
    long m = r0 + w * 16 + r + mh;
    if (m < nrows) {
#pragma unroll
      for (int nt = 0; nt < 8; ++nt) {
        int n = nt * 16 + nl;
        float v = acc[nt][r] + agts[m * DD + n];
        out[m * DD + n] = fmaxf(v, 0.f);
      }
    }
  }
}

// ---------------------------------------------------------------------------
// Host launcher
// ---------------------------------------------------------------------------
extern "C" void kernel_launch(void* const* d_in, const int* in_sizes, int n_in,
                              void* d_out, int out_size, void* d_ws, size_t ws_size,
                              hipStream_t stream) {
  const float* agts     = (const float*)d_in[0];
  const float* ctx      = (const float*)d_in[1];
  const float* agt_ctrs = (const float*)d_in[2];
  const float* ctx_ctrs = (const float*)d_in[3];
  const int*   hi       = (const int*)d_in[4];
  const int*   wi       = (const int*)d_in[5];
  const float* W_d0  = (const float*)d_in[6];
  const float* b_d0  = (const float*)d_in[7];
  const float* W_d1  = (const float*)d_in[8];
  const float* b_d1  = (const float*)d_in[9];
  const float* g_d1  = (const float*)d_in[10];
  const float* be_d1 = (const float*)d_in[11];
  const float* W_q   = (const float*)d_in[12];
  const float* b_q   = (const float*)d_in[13];
  const float* g_q   = (const float*)d_in[14];
  const float* be_q  = (const float*)d_in[15];
  const float* W_c   = (const float*)d_in[16];
  const float* b_c   = (const float*)d_in[17];
  const float* g_c   = (const float*)d_in[18];
  const float* be_c  = (const float*)d_in[19];
  const float* W_c2  = (const float*)d_in[20];
  const float* W_a   = (const float*)d_in[21];
  const float* g_n   = (const float*)d_in[22];
  const float* be_n  = (const float*)d_in[23];
  const float* W_l   = (const float*)d_in[24];
  const float* b_l   = (const float*)d_in[25];
  const float* g_l   = (const float*)d_in[26];
  const float* be_l  = (const float*)d_in[27];

  const int E     = in_sizes[4];
  const int nrows = in_sizes[0] / DD;   // N_AGT
  const int ncr   = in_sizes[1] / DD;   // N_CTX

  // Workspace layout (256B aligned).
  char* ws = (char*)d_ws;
  size_t off = 0;
  auto alignup = [](size_t x) { return (x + 255) & ~(size_t)255; };
  float* abuf = (float*)(ws + off);
  off = alignup(off + (size_t)nrows * DD * sizeof(float));
  _Float16* agtsh = (_Float16*)(ws + off); off = alignup(off + (size_t)nrows * DD * 2);
  _Float16* ctxh  = (_Float16*)(ws + off); off = alignup(off + (size_t)ncr * DD * 2);
  _Float16* W_d1h = (_Float16*)(ws + off); off = alignup(off + (size_t)DD * DD * 2);
  _Float16* W_qh  = (_Float16*)(ws + off); off = alignup(off + (size_t)DD * DD * 2);
  _Float16* W_ch  = (_Float16*)(ws + off); off = alignup(off + (size_t)3 * DD * DD * 2);
  _Float16* W_c2h = (_Float16*)(ws + off); off = alignup(off + (size_t)DD * DD * 2);
  _Float16* W_ah  = (_Float16*)(ws + off); off = alignup(off + (size_t)DD * DD * 2);
  _Float16* W_lh  = (_Float16*)(ws + off); off = alignup(off + (size_t)DD * DD * 2);

  const int nW  = DD * DD;
  const int nWc = 3 * DD * DD;
  const int nAg = nrows * DD;
  const int nCx = ncr * DD;
  convert_f16_kernel<<<(nAg + 255) / 256, 256, 0, stream>>>(agts, agtsh, nAg);
  convert_f16_kernel<<<(nCx + 255) / 256, 256, 0, stream>>>(ctx,  ctxh,  nCx);
  convert_f16_kernel<<<(nW  + 255) / 256, 256, 0, stream>>>(W_d1, W_d1h, nW);
  convert_f16_kernel<<<(nW  + 255) / 256, 256, 0, stream>>>(W_q,  W_qh,  nW);
  convert_f16_kernel<<<(nWc + 255) / 256, 256, 0, stream>>>(W_c,  W_ch,  nWc);
  convert_f16_kernel<<<(nW  + 255) / 256, 256, 0, stream>>>(W_c2, W_c2h, nW);
  convert_f16_kernel<<<(nW  + 255) / 256, 256, 0, stream>>>(W_a,  W_ah,  nW);
  convert_f16_kernel<<<(nW  + 255) / 256, 256, 0, stream>>>(W_l,  W_lh,  nW);

  base_kernel<<<(nrows + BM - 1) / BM, TPB, 0, stream>>>(agtsh, W_ah, abuf, nrows);

  edge_kernel<<<(E + BM - 1) / BM, TPB, 0, stream>>>(
      agtsh, ctxh, agt_ctrs, ctx_ctrs, hi, wi, E,
      W_d0, b_d0, b_d1, g_d1, be_d1,
      b_q, g_q, be_q, b_c, g_c, be_c,
      W_d1h, W_qh, W_ch, W_c2h, abuf);

  final_kernel<<<(nrows + BM - 1) / BM, TPB, 0, stream>>>(
      abuf, agts, g_n, be_n, W_lh, b_l, g_l, be_l, (float*)d_out, nrows);
}
